// onlyGate_51780125720667
// MI455X (gfx1250) — matrix operand order repaired and checked by reading, not compiled, not run on hardware
//
#include <hip/hip_runtime.h>

#define N_NODES 30000
#define N_EDGES 480000
#define HDIM    256
#define TETY    4
#define GRAPHS  64
#define CLASSES 10
#define NSTEPS  5
#define BN_EPS  1e-5f

typedef __attribute__((ext_vector_type(16))) __bf16       v16bf;
typedef __attribute__((ext_vector_type(8)))  float        v8f;
typedef __attribute__((ext_vector_type(4)))  unsigned int v4u;
typedef __attribute__((ext_vector_type(8)))  unsigned int v8u;

// ---------------------------------------------------------------- utilities

__global__ void zero_u32(unsigned int* __restrict__ p, int n) {
    int i = blockIdx.x * blockDim.x + threadIdx.x;
    if (i < n) p[i] = 0u;
}

__global__ void cvt_f32_bf16(const float* __restrict__ in, __bf16* __restrict__ out, int n) {
    int i = blockIdx.x * blockDim.x + threadIdx.x;
    if (i < n) out[i] = (__bf16)in[i];
}

// ------------------------------------------------------- WMMA GEMM (+bias)
// C[z][M, Ncols] = A_bf16[M,256] @ (W[z][Ncols,256])^T + bias[z][Ncols]
// Block: 256 threads = 8 waves (4 row-groups x 2 col-groups), tile 128x64.
// A tile double-buffered in LDS, filled with global_load_async_to_lds_b128
// (ASYNCcnt-tracked, in-order completion -> s_wait_asynccnt 2 pipelining).

__global__ __launch_bounds__(256)
void wmma_gemm_bias(const __bf16* __restrict__ A,
                    const __bf16* __restrict__ Wall,
                    const float*  __restrict__ biasAll,
                    float*        __restrict__ Call,
                    int Mrows, int Ncols)
{
    const int z = blockIdx.z;
    const __bf16* W   = Wall   + (size_t)z * Ncols * HDIM;
    const float* bias = biasAll + (size_t)z * Ncols;
    float*       C    = Call   + (size_t)z * (size_t)Mrows * Ncols;

    __shared__ __align__(16) __bf16 As[2 * 128 * 32];   // 2 x 8 KB A tiles
    const v4u* lds4 = (const v4u*)As;
    // LDS byte offset of As: low 32 bits of the generic address of a
    // local-addrspace object are the LDS offset on AMDGPU.
    const unsigned as_base = (unsigned)(size_t)(void*)As;

    const int tid   = threadIdx.x;
    const int lane  = tid & 31;
    const int wid   = tid >> 5;
    const int r0    = (wid & 3) * 32;   // wave row offset in tile
    const int c0    = (wid >> 2) * 32;  // wave col offset in tile
    const int blockRow = blockIdx.x * 128;
    const int blockCol = blockIdx.y * 64;

    const int m     = lane & 15;        // A-frag row within 16
    const int half  = lane >> 4;        // A-frag K-half select
    const int c     = lane & 15;        // B-frag column within 16
    const int khalf = lane >> 4;        // B-frag K-half select

    v8f acc00 = {}, acc01 = {}, acc10 = {}, acc11 = {};

    // async fill of one 128x32 bf16 A tile: 2 x b128 per thread
    auto issue_tile = [&](int kk, int buf) {
        #pragma unroll
        for (int i = 0; i < 2; ++i) {
            int idx  = tid + i * 256;           // enumerates (row, 16B seg)
            int row  = idx >> 2;
            int seg  = idx & 3;
            int grow = blockRow + row;
            if (grow >= Mrows) grow = Mrows - 1;   // clamp: rows never stored
            unsigned lds_addr = as_base + (unsigned)(buf * 8192 + idx * 16);
            unsigned long long gaddr = (unsigned long long)(const char*)A
                + ((unsigned long long)grow * (HDIM * 2) + (unsigned)(kk * 64 + seg * 16));
            asm volatile("global_load_async_to_lds_b128 %0, %1, off"
                         :: "v"(lds_addr), "v"(gaddr) : "memory");
        }
    };

    issue_tile(0, 0);

    for (int kk = 0; kk < 8; ++kk) {    // K = 256 in chunks of 32
        const int cur = kk & 1;
        if (kk < 7) {
            issue_tile(kk + 1, cur ^ 1);
            // 2 loads of the next tile in flight; wait for current tile only
            asm volatile("s_wait_asynccnt 2" ::: "memory");
        } else {
            asm volatile("s_wait_asynccnt 0" ::: "memory");
        }
        __syncthreads();                 // all waves' fills visible

        // A fragments (contiguous k = 16*half + s, matching B: two adjacent 16B chunks)
        const v4u* tile = lds4 + cur * 512;
        v8u ta0, ta1;
        ta0.lo = tile[(r0 + m) * 4 + 2 * half];
        ta0.hi = tile[(r0 + m) * 4 + 2 * half + 1];
        ta1.lo = tile[(r0 + 16 + m) * 4 + 2 * half];
        ta1.hi = tile[(r0 + 16 + m) * 4 + 2 * half + 1];
        v16bf a0 = __builtin_bit_cast(v16bf, ta0);
        v16bf a1 = __builtin_bit_cast(v16bf, ta1);

        // B fragments: B = W^T, so a B column = contiguous W row (32B chunk)
        const __bf16* wp0 = W + (size_t)(blockCol + c0 + c)      * HDIM + kk * 32 + khalf * 16;
        const __bf16* wp1 = W + (size_t)(blockCol + c0 + 16 + c) * HDIM + kk * 32 + khalf * 16;
        v16bf b0 = __builtin_bit_cast(v16bf, *(const v8u*)wp0);
        v16bf b1 = __builtin_bit_cast(v16bf, *(const v8u*)wp1);

        acc00 = __builtin_amdgcn_wmma_f32_16x16x32_bf16(false, a0, false, b0, (short)0, acc00, false, false);
        acc01 = __builtin_amdgcn_wmma_f32_16x16x32_bf16(false, a0, false, b1, (short)0, acc01, false, false);
        acc10 = __builtin_amdgcn_wmma_f32_16x16x32_bf16(false, a1, false, b0, (short)0, acc10, false, false);
        acc11 = __builtin_amdgcn_wmma_f32_16x16x32_bf16(false, a1, false, b1, (short)0, acc11, false, false);

        __syncthreads();                 // protect buffer reuse next iter
    }

    // epilogue: D VGPR j -> row j + 8*(lane/16), col lane%16
    const int n     = lane & 15;
    const int halfm = lane >> 4;
    const float bias0 = bias[blockCol + c0 + n];
    const float bias1 = bias[blockCol + c0 + 16 + n];
    #pragma unroll
    for (int j = 0; j < 8; ++j) {
        int row0 = blockRow + r0 + j + 8 * halfm;
        int row1 = row0 + 16;
        int col0 = blockCol + c0 + n;
        int col1 = col0 + 16;
        if (row0 < Mrows) {
            C[(size_t)row0 * Ncols + col0] = acc00[j] + bias0;
            C[(size_t)row0 * Ncols + col1] = acc01[j] + bias1;
        }
        if (row1 < Mrows) {
            C[(size_t)row1 * Ncols + col0] = acc10[j] + bias0;
            C[(size_t)row1 * Ncols + col1] = acc11[j] + bias1;
        }
    }
}

// --------------------------------------------------- edge gather + scatter
// a[dst[e]] += Wh[etype[e]][src[e]]  (float4 per thread, 4 global f32 atomics)

__global__ void edge_scatter(const float* __restrict__ Wh,
                             const int* __restrict__ src,
                             const int* __restrict__ dst,
                             const int* __restrict__ etype,
                             float* __restrict__ acc)
{
    long long gid = (long long)blockIdx.x * blockDim.x + threadIdx.x;
    int e  = (int)(gid >> 6);
    int c4 = (int)(gid & 63);
    if (e >= N_EDGES) return;
    int s = src[e], d = dst[e], t = etype[e];
    const float4 v = ((const float4*)(Wh + ((size_t)t * N_NODES + s) * HDIM))[c4];
    float* p = acc + (size_t)d * HDIM + c4 * 4;
    atomicAdd(p + 0, v.x);
    atomicAdd(p + 1, v.y);
    atomicAdd(p + 2, v.z);
    atomicAdd(p + 3, v.w);
}

// ----------------------------------------------------------- GRU cell fuse

__global__ void gru_update(const float* __restrict__ gi,
                           const float* __restrict__ gh,
                           float* __restrict__ h)
{
    int idx = blockIdx.x * blockDim.x + threadIdx.x;
    if (idx >= N_NODES * HDIM) return;
    int nrow = idx >> 8;
    int k    = idx & 255;
    size_t base = (size_t)nrow * 768;
    float i_r = gi[base + k], i_z = gi[base + 256 + k], i_n = gi[base + 512 + k];
    float h_r = gh[base + k], h_z = gh[base + 256 + k], h_n = gh[base + 512 + k];
    float r  = 1.f / (1.f + __expf(-(i_r + h_r)));
    float zz = 1.f / (1.f + __expf(-(i_z + h_z)));
    float nn = tanhf(i_n + r * h_n);
    h[idx] = (1.f - zz) * nn + zz * h[idx];
}

// --------------------------------------------- ELU + batch stats (2 pass BN)

__global__ void elu_stats(const float* __restrict__ h,
                          float* __restrict__ h2,
                          float* __restrict__ colsum,
                          float* __restrict__ colsq)
{
    int col = threadIdx.x;              // 256 threads == HDIM columns
    int r0  = blockIdx.x * 128;
    float s = 0.f, q = 0.f;
    for (int i = 0; i < 128; ++i) {
        int row = r0 + i;
        if (row >= N_NODES) break;
        float x = h[(size_t)row * HDIM + col];
        float e = x > 0.f ? x : (__expf(x) - 1.f);
        h2[(size_t)row * HDIM + col] = e;
        s += e; q += e * e;
    }
    atomicAdd(&colsum[col], s);
    atomicAdd(&colsq[col], q);
}

__global__ void bn_apply(float* __restrict__ h2,
                         const float* __restrict__ colsum,
                         const float* __restrict__ colsq,
                         const float* __restrict__ gamma,
                         const float* __restrict__ beta)
{
    int idx = blockIdx.x * blockDim.x + threadIdx.x;
    if (idx >= N_NODES * HDIM) return;
    int col = idx & 255;
    float mu  = colsum[col] * (1.f / N_NODES);
    float var = colsq[col] * (1.f / N_NODES) - mu * mu;
    h2[idx] = (h2[idx] - mu) * rsqrtf(var + BN_EPS) * gamma[col] + beta[col];
}

// ------------------------------------------- gate scores + segment softmax

__device__ __forceinline__ unsigned int enc_f32(float f) {
    unsigned int u = __float_as_uint(f);
    return (u & 0x80000000u) ? ~u : (u | 0x80000000u);
}
__device__ __forceinline__ float dec_f32(unsigned int u) {
    u = (u & 0x80000000u) ? (u & 0x7fffffffu) : ~u;
    return __uint_as_float(u);
}

__global__ void gate_kernel(const float* __restrict__ h2,
                            const float* __restrict__ gw,
                            const float* __restrict__ gb,
                            const int* __restrict__ n2g,
                            float* __restrict__ gate,
                            unsigned int* __restrict__ gmax_enc)
{
    int wid  = threadIdx.x >> 5;
    int lane = threadIdx.x & 31;
    int node = blockIdx.x * 8 + wid;
    if (node >= N_NODES) return;
    float s = 0.f;
    #pragma unroll
    for (int i = 0; i < 8; ++i) {
        int c = i * 32 + lane;
        s += h2[(size_t)node * HDIM + c] * gw[c];
    }
    #pragma unroll
    for (int off = 16; off > 0; off >>= 1)
        s += __shfl_down(s, off, 32);
    if (lane == 0) {
        float g = s + gb[0];
        gate[node] = g;
        atomicMax(&gmax_enc[n2g[node]], enc_f32(g));
    }
}

__global__ void softmax_num(const float* __restrict__ gate,
                            const int* __restrict__ n2g,
                            const unsigned int* __restrict__ gmax_enc,
                            float* __restrict__ e_arr,
                            float* __restrict__ denom)
{
    int i = blockIdx.x * blockDim.x + threadIdx.x;
    if (i >= N_NODES) return;
    int g = n2g[i];
    float ex = __expf(gate[i] - dec_f32(gmax_enc[g]));
    e_arr[i] = ex;
    atomicAdd(&denom[g], ex);
}

__global__ void pool_kernel(const float* __restrict__ h2,
                            const int* __restrict__ n2g,
                            const float* __restrict__ e_arr,
                            const float* __restrict__ denom,
                            float* __restrict__ h_g)
{
    long long gid = (long long)blockIdx.x * blockDim.x + threadIdx.x;
    int node = (int)(gid >> 6);
    int c4   = (int)(gid & 63);
    if (node >= N_NODES) return;
    int g = n2g[node];
    float alpha = e_arr[node] / denom[g];
    float4 v = ((const float4*)(h2 + (size_t)node * HDIM))[c4];
    float* p = h_g + (size_t)g * HDIM + c4 * 4;
    atomicAdd(p + 0, v.x * alpha);
    atomicAdd(p + 1, v.y * alpha);
    atomicAdd(p + 2, v.z * alpha);
    atomicAdd(p + 3, v.w * alpha);
}

// --------------------------------------------------------- tiny classifier

__global__ void mlp1(const float* __restrict__ h_g,
                     const float* __restrict__ W1,
                     const float* __restrict__ b1,
                     float* __restrict__ hid)
{
    int g = blockIdx.x, j = threadIdx.x;       // 64 blocks x 128 threads
    float s = b1[j];
    for (int k = 0; k < HDIM; ++k) s += h_g[g * HDIM + k] * W1[j * HDIM + k];
    hid[g * 128 + j] = fmaxf(s, 0.f);
}

__global__ void mlp2(const float* __restrict__ hid,
                     const float* __restrict__ W2,
                     const float* __restrict__ b2,
                     float* __restrict__ out)
{
    int idx = threadIdx.x;                     // 1 block x 640 threads
    if (idx >= GRAPHS * CLASSES) return;
    int g = idx / CLASSES, c = idx % CLASSES;
    float s = b2[c];
    for (int k = 0; k < 128; ++k) s += hid[g * 128 + k] * W2[c * 128 + k];
    out[idx] = s;
}

// ---------------------------------------------------------------- launcher

extern "C" void kernel_launch(void* const* d_in, const int* in_sizes, int n_in,
                              void* d_out, int out_size, void* d_ws, size_t ws_size,
                              hipStream_t stream) {
    const float* feat   = (const float*)d_in[0];
    const int*   src    = (const int*)d_in[1];
    const int*   dst    = (const int*)d_in[2];
    const int*   etype  = (const int*)d_in[3];
    const int*   n2g    = (const int*)d_in[4];
    const float* W_msg  = (const float*)d_in[5];
    const float* b_msg  = (const float*)d_in[6];
    const float* w_ih   = (const float*)d_in[7];
    const float* w_hh   = (const float*)d_in[8];
    const float* b_ih   = (const float*)d_in[9];
    const float* b_hh   = (const float*)d_in[10];
    const float* bn_g   = (const float*)d_in[11];
    const float* bn_b   = (const float*)d_in[12];
    const float* gate_w = (const float*)d_in[13];
    const float* gate_b = (const float*)d_in[14];
    const float* W1     = (const float*)d_in[15];
    const float* b1     = (const float*)d_in[16];
    const float* W2     = (const float*)d_in[17];
    const float* b2     = (const float*)d_in[18];
    float* out = (float*)d_out;

    char* ws = (char*)d_ws;
    size_t off = 0;
    auto alloc = [&](size_t bytes) -> char* {
        char* p = ws + off;
        off += (bytes + 255) & ~(size_t)255;
        return p;
    };

    const size_t NH = (size_t)N_NODES * HDIM;
    float*  h       = (float*)alloc(NH * 4);
    __bf16* hbf     = (__bf16*)alloc(NH * 2);
    float*  a       = (float*)alloc(NH * 4);
    __bf16* abf     = (__bf16*)alloc(NH * 2);
    float*  Wh      = (float*)alloc((size_t)TETY * NH * 4);
    float*  gi      = (float*)alloc((size_t)N_NODES * 768 * 4);
    float*  gh      = (float*)alloc((size_t)N_NODES * 768 * 4);
    float*  h2      = (float*)alloc(NH * 4);
    __bf16* wmsg_bf = (__bf16*)alloc((size_t)TETY * HDIM * HDIM * 2);
    __bf16* wih_bf  = (__bf16*)alloc((size_t)768 * HDIM * 2);
    __bf16* whh_bf  = (__bf16*)alloc((size_t)768 * HDIM * 2);
    float*  colsum  = (float*)alloc(HDIM * 4);
    float*  colsq   = (float*)alloc(HDIM * 4);
    float*  gate    = (float*)alloc((size_t)N_NODES * 4);
    float*  e_arr   = (float*)alloc((size_t)N_NODES * 4);
    unsigned int* gmax = (unsigned int*)alloc(GRAPHS * 4);
    float*  denom   = (float*)alloc(GRAPHS * 4);
    float*  h_g     = (float*)alloc((size_t)GRAPHS * HDIM * 4);
    float*  hid     = (float*)alloc((size_t)GRAPHS * 128 * 4);

    // h = feat ; weights -> bf16 (done every call: deterministic, capture-safe)
    hipMemcpyAsync(h, feat, NH * 4, hipMemcpyDeviceToDevice, stream);
    cvt_f32_bf16<<<(TETY * HDIM * HDIM + 255) / 256, 256, 0, stream>>>(W_msg, wmsg_bf, TETY * HDIM * HDIM);
    cvt_f32_bf16<<<(768 * HDIM + 255) / 256, 256, 0, stream>>>(w_ih, wih_bf, 768 * HDIM);
    cvt_f32_bf16<<<(768 * HDIM + 255) / 256, 256, 0, stream>>>(w_hh, whh_bf, 768 * HDIM);

    const int mBlocks = (N_NODES + 127) / 128;   // 235
    for (int step = 0; step < NSTEPS; ++step) {
        cvt_f32_bf16<<<(int)(NH / 256), 256, 0, stream>>>(h, hbf, (int)NH);
        zero_u32<<<(int)(NH / 256), 256, 0, stream>>>((unsigned int*)a, (int)NH);
        // Wh[t] = h @ W_msg[t]^T + b_msg[t] for t=0..3 (blockIdx.z)
        wmma_gemm_bias<<<dim3(mBlocks, HDIM / 64, TETY), 256, 0, stream>>>(
            hbf, wmsg_bf, b_msg, Wh, N_NODES, HDIM);
        // a[dst] += Wh[etype][src]
        edge_scatter<<<(int)(((long long)N_EDGES * 64) / 256), 256, 0, stream>>>(
            Wh, src, dst, etype, a);
        cvt_f32_bf16<<<(int)(NH / 256), 256, 0, stream>>>(a, abf, (int)NH);
        // gi = a @ w_ih^T + b_ih ; gh = h @ w_hh^T + b_hh
        wmma_gemm_bias<<<dim3(mBlocks, 768 / 64, 1), 256, 0, stream>>>(
            abf, wih_bf, b_ih, gi, N_NODES, 768);
        wmma_gemm_bias<<<dim3(mBlocks, 768 / 64, 1), 256, 0, stream>>>(
            hbf, whh_bf, b_hh, gh, N_NODES, 768);
        gru_update<<<(int)(NH / 256), 256, 0, stream>>>(gi, gh, h);
    }

    // ELU + BatchNorm (training-mode batch stats)
    zero_u32<<<1, 256, 0, stream>>>((unsigned int*)colsum, HDIM);
    zero_u32<<<1, 256, 0, stream>>>((unsigned int*)colsq, HDIM);
    elu_stats<<<mBlocks, HDIM, 0, stream>>>(h, h2, colsum, colsq);
    bn_apply<<<(int)(NH / 256), 256, 0, stream>>>(h2, colsum, colsq, bn_g, bn_b);

    // Global attention pooling
    zero_u32<<<1, 256, 0, stream>>>(gmax, GRAPHS);
    zero_u32<<<1, 256, 0, stream>>>((unsigned int*)denom, GRAPHS);
    zero_u32<<<(GRAPHS * HDIM) / 256, 256, 0, stream>>>((unsigned int*)h_g, GRAPHS * HDIM);
    gate_kernel<<<(N_NODES + 7) / 8, 256, 0, stream>>>(h2, gate_w, gate_b, n2g, gate, gmax);
    softmax_num<<<(N_NODES + 255) / 256, 256, 0, stream>>>(gate, n2g, gmax, e_arr, denom);
    pool_kernel<<<(int)(((long long)N_NODES * 64) / 256), 256, 0, stream>>>(h2, n2g, e_arr, denom, h_g);

    // classifier
    mlp1<<<GRAPHS, 128, 0, stream>>>(h_g, W1, b1, hid);
    mlp2<<<1, 640, 0, stream>>>(hid, W2, b2, out);
}
